// ReccurentGNN_87875030876658
// MI455X (gfx1250) — matrix operand order repaired and checked by reading, not compiled, not run on hardware
//
#include <hip/hip_runtime.h>

#define N_NODES 50000
#define N_EDGES 1600000
#define CHUNK   262144
#define INV_RADIUS (1.0f/0.075f)

typedef __attribute__((ext_vector_type(16))) _Float16 v16h;
typedef __attribute__((ext_vector_type(8)))  _Float16 v8h;
typedef __attribute__((ext_vector_type(8)))  float    v8f;
typedef __attribute__((ext_vector_type(4)))  int      v4i;

#define AS1 __attribute__((address_space(1)))
#define AS3 __attribute__((address_space(3)))

#if defined(__gfx1250__) && __has_builtin(__builtin_amdgcn_global_load_async_to_lds_b128)
#define USE_ASYNC_LDS 1
#else
#define USE_ASYNC_LDS 0
#endif

__device__ __forceinline__ void wait_async_done() {
#if USE_ASYNC_LDS
#if __has_builtin(__builtin_amdgcn_s_wait_asynccnt)
  __builtin_amdgcn_s_wait_asynccnt(0);
#else
  asm volatile("s_wait_asynccnt 0x0" ::: "memory");
#endif
#endif
}

// ---------------------------------------------------------------------------
// Small elementwise kernels
// ---------------------------------------------------------------------------
__global__ void convert_w(const float* __restrict__ W, _Float16* __restrict__ Wp,
                          int Nr, int Kr, int Kp, int total) {
  int i = blockIdx.x * 256 + threadIdx.x;
  if (i >= total) return;
  int n = i / Kp, k = i - n * Kp;
  float v = (n < Nr && k < Kr) ? W[(size_t)n * Kr + k] : 0.0f;
  Wp[i] = (_Float16)v;
}

__global__ void zero_f32(float* __restrict__ p, int n) {
  int i = blockIdx.x * 256 + threadIdx.x;
  if (i < n) p[i] = 0.0f;
}

__global__ void f32_to_f16(const float* __restrict__ a, _Float16* __restrict__ b, int n) {
  int i = blockIdx.x * 256 + threadIdx.x;
  if (i < n) b[i] = (_Float16)a[i];
}

__global__ void build_node_feats(const float* __restrict__ x, _Float16* __restrict__ nf) {
  int i = blockIdx.x * 256 + threadIdx.x;
  if (i >= N_NODES) return;
  const float* xi = x + (size_t)i * 5;
  _Float16* o = nf + (size_t)i * 32;
  o[0] = (_Float16)xi[0]; o[1] = (_Float16)xi[1];
  o[2] = (_Float16)xi[2]; o[3] = (_Float16)xi[3];
  _Float16 t = (_Float16)xi[4];
  o[4] = t; o[5] = t; o[6] = t; o[7] = t;
  #pragma unroll
  for (int k = 8; k < 32; ++k) o[k] = (_Float16)0.0f;
}

__global__ void build_edge_feats(const float* __restrict__ x,
                                 const int* __restrict__ src, const int* __restrict__ dst,
                                 long long base, int count, _Float16* __restrict__ ef) {
  int i = blockIdx.x * 256 + threadIdx.x;
  if (i >= count) return;
  long long e = base + i;
  int s = src[e], d = dst[e];
  const float* xi = x + (size_t)d * 5;
  const float* xj = x + (size_t)s * 5;
  float dx = xi[0] - xj[0], dy = xi[1] - xj[1];
  float rx = dx * INV_RADIUS, ry = dy * INV_RADIUS;
  float r  = sqrtf(dx * dx + dy * dy) * INV_RADIUS;
  _Float16* o = ef + (size_t)i * 32;
  o[0] = (_Float16)rx; o[1] = (_Float16)ry; o[2] = (_Float16)r;
  o[3] = (_Float16)xi[2]; o[4] = (_Float16)xi[3];
  o[5] = (_Float16)xj[2]; o[6] = (_Float16)xj[3];
  _Float16 t = (_Float16)xi[4];
  o[7] = t; o[8] = t; o[9] = t; o[10] = t;
  #pragma unroll
  for (int k = 11; k < 32; ++k) o[k] = (_Float16)0.0f;
}

// ---------------------------------------------------------------------------
// WMMA GEMM: Y[M x Npad] = A[M x Kp] * W^T  (+ bias, epilogue variants)
// One wave = 16(M) x (NF*16)(N); block = 8 waves = 128 rows; grid.y == 1 for
// every layer (full N per block), so each A row is streamed from HBM once.
// Weight tile (NF*16 rows x 32 K, f16) staged in LDS per k-step, via
// GLOBAL_LOAD_ASYNC_TO_LDS_B128 when the toolchain exposes it.
//
// ASRC: 0 = dense A (lda)           — plain activations
//       1 = concat2 [node|aggr]     — K=128, row stride 64 each
//       2 = concat3 [edge|h_i|h_j]  — K=192, gathers via dst/src indices
// EPI : 0 = ReLU -> f16   1 = f16   2 = f32+f16 (node embedding)
//       3 = message: edge += msg (f16) + atomicAdd into aggr[dst]
//       4 = node:    node_f32 += out, refresh f16 copy
//       5 = final prediction (Nreal=2, f32 -> d_out)
// ---------------------------------------------------------------------------
struct GemmArgs {
  const _Float16* A;      long long lda;
  const _Float16* W;      const float* bias;
  int M, Kp, Npad, Nreal;
  const _Float16* segA0;  const _Float16* segA1; const _Float16* segA2;
  const int* srcIdx;      const int* dstIdx;     long long rowBase;
  _Float16* outH;         long long ldoH;
  float*    outF;                          // EPI 5
  float*    aggr;                          // EPI 3
  _Float16* edgeH;                         // EPI 3 (edge master, stride 64)
  float*    nodeF;        _Float16* nodeH; // EPI 2/4
};

template<int ASRC, int EPI, int NF>
__global__ __launch_bounds__(256) void gemm_wmma(GemmArgs g) {
  __shared__ _Float16 lw[NF * 16 * 32];     // up to 12 KB (NF=12)
  const int tid  = threadIdx.x;
  const int lane = tid & 31;
  const int wave = tid >> 5;
  const int m0   = (blockIdx.x * 8 + wave) * 16;
  const int n0   = blockIdx.y * (NF * 16);
  const int hi   = lane >> 4;      // 0: lanes 0-15, 1: lanes 16-31
  const int lm   = lane & 15;

  int mA = m0 + lm; if (mA >= g.M) mA = g.M - 1;   // clamp; dead rows never stored
  long long erow = 0; int di = 0, si = 0;
  if (ASRC == 2) { erow = g.rowBase + mA; si = g.srcIdx[erow]; di = g.dstIdx[erow]; }
  const _Float16* aRow = (ASRC == 0) ? (g.A + (size_t)mA * g.lda) : (const _Float16*)0;

  v8f acc[NF] = {};

  for (int k0 = 0; k0 < g.Kp; k0 += 32) {
    __syncthreads();
    // Stage W tile [NF*16 x 32] into LDS; rows always valid (N padded to 64).
    {
      const int col = (tid & 3) << 3;
      #pragma unroll
      for (int row = tid >> 2; row < NF * 16; row += 64) {
        const _Float16* gp = g.W + (size_t)(n0 + row) * g.Kp + k0 + col;
        _Float16* lp = lw + row * 32 + col;
#if USE_ASYNC_LDS
        __builtin_amdgcn_global_load_async_to_lds_b128(
            (AS1 v4i*)gp, (AS3 v4i*)lp, /*imm_offset=*/0, /*cpol=*/0);
#else
        *(v8h*)lp = *(const v8h*)gp;
#endif
      }
    }
    wait_async_done();
    __syncthreads();

    // A fragment (16x32 f16): lanes 0-15 hold K k0+0..15 of row M=lm,
    // lanes 16-31 hold K k0+16..31 of the same row.
    const _Float16* rp; int kk;
    if (ASRC == 0)      { rp = aRow; kk = k0; }
    else if (ASRC == 1) { int seg = k0 >> 6; kk = k0 & 63;
                          rp = (seg == 0 ? g.segA0 : g.segA1) + (size_t)mA * 64; }
    else                { int seg = k0 >> 6; kk = k0 & 63;
                          rp = (seg == 0) ? g.segA0 + (size_t)erow * 64
                             : (seg == 1) ? g.segA1 + (size_t)di * 64
                                          : g.segA2 + (size_t)si * 64; }
    union { v16h v; v8h h[2]; } ua;
    int ko = kk + (hi << 4);
    ua.h[0] = *(const v8h*)(rp + ko);
    ua.h[1] = *(const v8h*)(rp + ko + 8);

    #pragma unroll
    for (int j = 0; j < NF; ++j) {
      // B fragment (32x16): lane = column N, 16 contiguous K per half-wave
      const _Float16* bp = lw + (j * 16 + lm) * 32 + (hi << 4);
      v16h b = *(const v16h*)bp;
      acc[j] = __builtin_amdgcn_wmma_f32_16x16x32_f16(
          /*neg_a=*/false, ua.v, /*neg_b=*/false, b,
          /*c_mod=*/(short)0, acc[j], /*reuse_a=*/false, /*reuse_b=*/false);
    }
  }

  // Epilogue: C/D layout — VGPR r: M = m0 + r + 8*hi, N = n0 + 16*j + lm
  #pragma unroll
  for (int j = 0; j < NF; ++j) {
    int n = n0 + j * 16 + lm;
    float bb = (n < g.Nreal) ? g.bias[n] : 0.0f;
    #pragma unroll
    for (int r = 0; r < 8; ++r) {
      int m = m0 + r + (hi << 3);
      if (m >= g.M) continue;
      float v = acc[j][r] + bb;
      if (EPI == 0) {
        v = v > 0.0f ? v : 0.0f;
        g.outH[(size_t)m * g.ldoH + n] = (_Float16)v;
      } else if (EPI == 1) {
        g.outH[(size_t)m * g.ldoH + n] = (_Float16)v;
      } else if (EPI == 2) {
        size_t o = (size_t)m * 64 + n;
        g.nodeF[o] = v; g.nodeH[o] = (_Float16)v;
      } else if (EPI == 3) {
        long long e = g.rowBase + m;
        size_t o = (size_t)e * 64 + n;
        float ev = (float)g.edgeH[o] + v;      // edge = edge + msg
        g.edgeH[o] = (_Float16)ev;
        int dd = g.dstIdx[e];
        atomicAdd(g.aggr + (size_t)dd * 64 + n, v);  // segment_sum
      } else if (EPI == 4) {
        size_t o = (size_t)m * 64 + n;
        float nv = g.nodeF[o] + v;             // node = node + mlp(...)
        g.nodeF[o] = nv; g.nodeH[o] = (_Float16)nv;
      } else {
        if (n < g.Nreal) g.outF[(size_t)m * g.Nreal + n] = v;
      }
    }
  }
}

// ---------------------------------------------------------------------------
// Host side
// ---------------------------------------------------------------------------
static void gemmL(int asrc, int epi, int nf, const GemmArgs& g, hipStream_t s) {
  dim3 grid((g.M + 127) / 128, g.Npad / (nf * 16));
  dim3 blk(256);
#define CASE(A, E, N) \
  if (asrc == A && epi == E && nf == N) { gemm_wmma<A, E, N><<<grid, blk, 0, s>>>(g); return; }
  CASE(0, 0, 4)  CASE(0, 0, 8)  CASE(0, 0, 12)
  CASE(0, 1, 4)  CASE(0, 2, 4)  CASE(0, 3, 4)
  CASE(0, 4, 4)  CASE(0, 5, 4)
  CASE(1, 0, 8)  CASE(2, 0, 12)
#undef CASE
}

struct WDim { int Nr, Kr, Np, Kp; };

extern "C" void kernel_launch(void* const* d_in, const int* in_sizes, int n_in,
                              void* d_out, int out_size, void* d_ws, size_t ws_size,
                              hipStream_t stream) {
  const float* x  = (const float*)d_in[0];
  const int*   ei = (const int*)d_in[1];
  const int*   src = ei;             // edge_index[0]
  const int*   dst = ei + N_EDGES;   // edge_index[1]

  // 33 (W, b) pairs in setup_inputs() insertion order
  const float* Wsrc[33]; const float* Bias[33];
  for (int i = 0; i < 33; ++i) { Wsrc[i] = (const float*)d_in[2 + 2 * i];
                                 Bias[i] = (const float*)d_in[3 + 2 * i]; }
  WDim wd[33]; int wi = 0;
  wd[wi++] = {64, 8, 64, 32};  wd[wi++] = {64, 64, 64, 64}; wd[wi++] = {64, 64, 64, 64};
  wd[wi++] = {64, 11, 64, 32}; wd[wi++] = {64, 64, 64, 64}; wd[wi++] = {64, 64, 64, 64};
  for (int l = 0; l < 4; ++l) {
    wd[wi++] = {192, 192, 192, 192}; wd[wi++] = {192, 192, 192, 192}; wd[wi++] = {64, 192, 64, 192};
    wd[wi++] = {128, 128, 128, 128}; wd[wi++] = {128, 128, 128, 128}; wd[wi++] = {64, 128, 64, 128};
  }
  wd[wi++] = {64, 64, 64, 64}; wd[wi++] = {64, 64, 64, 64}; wd[wi++] = {2, 64, 64, 64};

  // Workspace carve-up (256B aligned)
  char* base = (char*)d_ws; size_t off = 0;
  auto carve = [&](size_t bytes) -> void* {
    void* p = base + off; off = (off + bytes + 255) & ~(size_t)255; return p;
  };
  _Float16* wp[33];
  for (int i = 0; i < 33; ++i) wp[i] = (_Float16*)carve((size_t)wd[i].Np * wd[i].Kp * 2);
  _Float16* nf_bf   = (_Float16*)carve((size_t)N_NODES * 32 * 2);
  _Float16* node_bf = (_Float16*)carve((size_t)N_NODES * 64 * 2);
  float*    node_f  = (float*)   carve((size_t)N_NODES * 64 * 4);
  float*    aggr_f  = (float*)   carve((size_t)N_NODES * 64 * 4);
  _Float16* aggr_bf = (_Float16*)carve((size_t)N_NODES * 64 * 2);
  _Float16* t_n1    = (_Float16*)carve((size_t)N_NODES * 128 * 2);
  _Float16* t_n2    = (_Float16*)carve((size_t)N_NODES * 128 * 2);
  _Float16* edge_bf = (_Float16*)carve((size_t)N_EDGES * 64 * 2);
  _Float16* ef_bf   = (_Float16*)carve((size_t)CHUNK * 32 * 2);
  _Float16* t_e1    = (_Float16*)carve((size_t)CHUNK * 192 * 2);
  _Float16* t_e2    = (_Float16*)carve((size_t)CHUNK * 192 * 2);
  if (off > ws_size) return;  // workspace too small; nothing safe to do

  // 1) weights fp32 -> f16 (zero-padded to Np x Kp)
  for (int i = 0; i < 33; ++i) {
    int tot = wd[i].Np * wd[i].Kp;
    convert_w<<<(tot + 255) / 256, 256, 0, stream>>>(Wsrc[i], wp[i], wd[i].Nr, wd[i].Kr, wd[i].Kp, tot);
  }

  // 2) node features + node embedding MLP [8->64->64->64]
  build_node_feats<<<(N_NODES + 255) / 256, 256, 0, stream>>>(x, nf_bf);
  GemmArgs g = {};
  g.srcIdx = src; g.dstIdx = dst;
  g.M = N_NODES; g.A = nf_bf; g.lda = 32; g.W = wp[0]; g.bias = Bias[0];
  g.Kp = 32; g.Npad = 64; g.Nreal = 64; g.outH = t_n1; g.ldoH = 64;
  gemmL(0, 0, 4, g, stream);
  g.A = t_n1; g.lda = 64; g.W = wp[1]; g.bias = Bias[1]; g.Kp = 64; g.outH = t_n2;
  gemmL(0, 0, 4, g, stream);
  g.A = t_n2; g.W = wp[2]; g.bias = Bias[2]; g.nodeF = node_f; g.nodeH = node_bf;
  gemmL(0, 2, 4, g, stream);

  // 3) edge features + edge embedding MLP [11->64->64->64], chunked
  for (long long cb = 0; cb < N_EDGES; cb += CHUNK) {
    int cnt = (int)((N_EDGES - cb) < CHUNK ? (N_EDGES - cb) : CHUNK);
    build_edge_feats<<<(cnt + 255) / 256, 256, 0, stream>>>(x, src, dst, cb, cnt, ef_bf);
    g = GemmArgs{}; g.srcIdx = src; g.dstIdx = dst; g.M = cnt;
    g.A = ef_bf; g.lda = 32; g.W = wp[3]; g.bias = Bias[3];
    g.Kp = 32; g.Npad = 64; g.Nreal = 64; g.outH = t_e1; g.ldoH = 64;
    gemmL(0, 0, 4, g, stream);
    g.A = t_e1; g.lda = 64; g.W = wp[4]; g.bias = Bias[4]; g.Kp = 64; g.outH = t_e2;
    gemmL(0, 0, 4, g, stream);
    g.A = t_e2; g.W = wp[5]; g.bias = Bias[5]; g.outH = edge_bf + cb * 64;
    gemmL(0, 1, 4, g, stream);
  }

  // 4) message-passing layers
  for (int l = 0; l < 4; ++l) {
    int we = 6 + 6 * l, wn = 9 + 6 * l;
    zero_f32<<<(N_NODES * 64 + 255) / 256, 256, 0, stream>>>(aggr_f, N_NODES * 64);

    for (long long cb = 0; cb < N_EDGES; cb += CHUNK) {
      int cnt = (int)((N_EDGES - cb) < CHUNK ? (N_EDGES - cb) : CHUNK);
      // edge MLP L0: concat [edge | h_i(dst) | h_j(src)] (K=192) -> 192, ReLU
      g = GemmArgs{}; g.srcIdx = src; g.dstIdx = dst; g.rowBase = cb; g.M = cnt;
      g.segA0 = edge_bf; g.segA1 = node_bf; g.segA2 = node_bf;
      g.W = wp[we]; g.bias = Bias[we]; g.Kp = 192; g.Npad = 192; g.Nreal = 192;
      g.outH = t_e1; g.ldoH = 192;
      gemmL(2, 0, 12, g, stream);
      // L1: 192 -> 192, ReLU
      g.A = t_e1; g.lda = 192; g.W = wp[we + 1]; g.bias = Bias[we + 1]; g.outH = t_e2;
      gemmL(0, 0, 12, g, stream);
      // L2: 192 -> 64 message; edge += msg, atomic scatter into aggr[dst]
      g.A = t_e2; g.W = wp[we + 2]; g.bias = Bias[we + 2];
      g.Npad = 64; g.Nreal = 64; g.edgeH = edge_bf; g.aggr = aggr_f;
      gemmL(0, 3, 4, g, stream);
    }
    f32_to_f16<<<(N_NODES * 64 + 255) / 256, 256, 0, stream>>>(aggr_f, aggr_bf, N_NODES * 64);

    // node MLP L0: concat [node | aggr] (K=128) -> 128, ReLU
    g = GemmArgs{}; g.srcIdx = src; g.dstIdx = dst; g.M = N_NODES;
    g.segA0 = node_bf; g.segA1 = aggr_bf;
    g.W = wp[wn]; g.bias = Bias[wn]; g.Kp = 128; g.Npad = 128; g.Nreal = 128;
    g.outH = t_n1; g.ldoH = 128;
    gemmL(1, 0, 8, g, stream);
    g.A = t_n1; g.lda = 128; g.W = wp[wn + 1]; g.bias = Bias[wn + 1]; g.outH = t_n2;
    gemmL(0, 0, 8, g, stream);
    // L2: 128 -> 64; node = node + out (f32 master + f16 copy)
    g.A = t_n2; g.W = wp[wn + 2]; g.bias = Bias[wn + 2];
    g.Npad = 64; g.Nreal = 64; g.nodeF = node_f; g.nodeH = node_bf;
    gemmL(0, 4, 4, g, stream);
  }

  // 5) node_out MLP [64->64->64->2]
  g = GemmArgs{}; g.srcIdx = src; g.dstIdx = dst; g.M = N_NODES;
  g.A = node_bf; g.lda = 64; g.W = wp[30]; g.bias = Bias[30];
  g.Kp = 64; g.Npad = 64; g.Nreal = 64; g.outH = t_n1; g.ldoH = 64;
  gemmL(0, 0, 4, g, stream);
  g.A = t_n1; g.W = wp[31]; g.bias = Bias[31]; g.outH = t_n2;
  gemmL(0, 0, 4, g, stream);
  g.A = t_n2; g.W = wp[32]; g.bias = Bias[32]; g.Nreal = 2; g.outF = (float*)d_out;
  gemmL(0, 5, 4, g, stream);
}
